// SelfAttention_22308060135460
// MI455X (gfx1250) — compile-verified, matrix operand order checked by reading, not run on hardware
//
#include <hip/hip_runtime.h>
#include <hip/hip_bf16.h>

typedef __attribute__((ext_vector_type(16))) __bf16 v16bf;
typedef __attribute__((ext_vector_type(8)))  float  v8f;

// ---------------------------------------------------------------------------
// fp32 -> bf16 (round to nearest even)
__device__ __forceinline__ unsigned short f2bf(float x) {
  unsigned int u = __float_as_uint(x);
  u += 0x7FFFu + ((u >> 16) & 1u);
  return (unsigned short)(u >> 16);
}

// Load a 16x32 bf16 fragment (CDNA5 ISA 7.12.2 16-bit A layout; also used for
// B as "rows of B^T"). Lane l: row = l%16, K-chunks [base, base+8) and
// [base+16, base+24) with base = (l/16)*8. Two 16-byte loads per lane.
__device__ __forceinline__ v16bf load_frag(const unsigned short* p, int ld) {
  const int lane = threadIdx.x & 31;
  const unsigned short* r = p + (size_t)(lane & 15) * ld + ((lane >> 4) << 3);
  v16bf f;
  uint4* fq = reinterpret_cast<uint4*>(&f);
  fq[0] = *reinterpret_cast<const uint4*>(r);
  fq[1] = *reinterpret_cast<const uint4*>(r + 16);
  return f;
}

#define WMMA_BF16(a, b, c) \
  __builtin_amdgcn_wmma_f32_16x16x32_bf16(false, (a), false, (b), (short)0, (c), false, false)

// ---------------------------------------------------------------------------
// Precision conversion kernels
__global__ void cvt_bf16(const float* __restrict__ src,
                         unsigned short* __restrict__ dst, int n) {
  int i = blockIdx.x * blockDim.x + threadIdx.x;
  if (i < n) dst[i] = f2bf(src[i]);
}

// src [rows][cols] fp32 -> dst [cols][rows] bf16 (store W^T so B-fragments
// become contiguous-chunk loads)
__global__ void cvt_bf16_t(const float* __restrict__ src,
                           unsigned short* __restrict__ dst, int rows, int cols) {
  int i = blockIdx.x * blockDim.x + threadIdx.x;
  if (i < rows * cols) {
    int n = i / rows, k = i - n * rows;
    dst[i] = f2bf(src[(size_t)k * cols + n]);
  }
}

// ---------------------------------------------------------------------------
// QKV projection: C[8192,1536] = X[8192,512] @ Wqkv + b, scattered into
// Q,K as bf16 [b,h,n,64] and V as bf16 [b,h,64,n] (transposed for flash PV).
// Block = 128 threads (4 waves, 2x2), block tile 64x64, wave tile 32x32.
__global__ void __launch_bounds__(128)
gemm_qkv(const unsigned short* __restrict__ A,   // x bf16 [8192][512]
         const unsigned short* __restrict__ Bt,  // WqkvT bf16 [1536][512]
         const float* __restrict__ bias,         // [1536]
         unsigned short* __restrict__ Qb, unsigned short* __restrict__ Kb,
         unsigned short* __restrict__ Vt) {
  const int K = 512;
  const int lane = threadIdx.x & 31;
  const int wv = threadIdx.x >> 5;
  const int m0 = blockIdx.y * 64 + (wv >> 1) * 32;
  const int n0 = blockIdx.x * 64 + (wv & 1) * 32;

  v8f c00 = {}, c01 = {}, c10 = {}, c11 = {};
  for (int kt = 0; kt < K; kt += 32) {
    v16bf a0 = load_frag(A + (size_t)m0 * K + kt, K);
    v16bf a1 = load_frag(A + (size_t)(m0 + 16) * K + kt, K);
    v16bf b0 = load_frag(Bt + (size_t)n0 * K + kt, K);
    v16bf b1 = load_frag(Bt + (size_t)(n0 + 16) * K + kt, K);
    c00 = WMMA_BF16(a0, b0, c00);
    c01 = WMMA_BF16(a0, b1, c01);
    c10 = WMMA_BF16(a1, b0, c10);
    c11 = WMMA_BF16(a1, b1, c11);
  }

  const int half = lane >> 4;
#pragma unroll
  for (int cr = 0; cr < 2; ++cr) {
#pragma unroll
    for (int cc = 0; cc < 2; ++cc) {
      const v8f& c = cr ? (cc ? c11 : c10) : (cc ? c01 : c00);
      int ng = n0 + cc * 16 + (lane & 15);
      float bv = bias[ng];
      int sect = ng >> 9;      // 0=Q 1=K 2=V
      int f = ng & 511;
      int h = f >> 6, d = f & 63;
#pragma unroll
      for (int v = 0; v < 8; ++v) {
        int mg = m0 + cr * 16 + v + (half << 3);  // token index
        unsigned short bw = f2bf(c[v] + bv);
        int b = mg >> 12, n = mg & 4095;
        int bh = b * 8 + h;
        if (sect == 0)      Qb[((size_t)(bh * 4096 + n) << 6) + d] = bw;
        else if (sect == 1) Kb[((size_t)(bh * 4096 + n) << 6) + d] = bw;
        else                Vt[((size_t)(bh * 64 + d) << 12) + n] = bw;
      }
    }
  }
}

// ---------------------------------------------------------------------------
// Flash attention: grid (4096/64 qtiles, 16 bh), block 128 (4 waves).
// Each wave owns 16 q rows, streams 32-key blocks, online softmax in VGPRs.
__global__ void __launch_bounds__(128)
flash_attn(const unsigned short* __restrict__ Qb,
           const unsigned short* __restrict__ Kb,
           const unsigned short* __restrict__ Vt,
           unsigned short* __restrict__ attnb) {
  __shared__ __align__(16) unsigned short pTile[4][16][40];  // padded rows

  const int lane = threadIdx.x & 31;
  const int wv = threadIdx.x >> 5;
  const int half = lane >> 4;
  const int bh = blockIdx.y;
  const int q0 = blockIdx.x * 64 + wv * 16;

  const unsigned short* Qp = Qb + ((size_t)bh * 4096 + q0) * 64;
  const unsigned short* Kp = Kb + (size_t)bh * 4096 * 64;
  const unsigned short* Vp = Vt + (size_t)bh * 64 * 4096;

  v16bf qf0 = load_frag(Qp, 64);       // d 0..31
  v16bf qf1 = load_frag(Qp + 32, 64);  // d 32..63

  v8f o0 = {}, o1 = {}, o2 = {}, o3 = {};
  float mrow[8], lrow[8];
#pragma unroll
  for (int v = 0; v < 8; ++v) { mrow[v] = -1e30f; lrow[v] = 0.0f; }

  for (int kb = 0; kb < 4096; kb += 32) {
    // S = (Q K^T) * 1/sqrt(64): two 16x16 score accs (keys kb.. / kb+16..)
    v8f s0 = {}, s1 = {};
    v16bf k00 = load_frag(Kp + (size_t)kb * 64, 64);
    v16bf k10 = load_frag(Kp + (size_t)(kb + 16) * 64, 64);
    s0 = WMMA_BF16(qf0, k00, s0);
    s1 = WMMA_BF16(qf0, k10, s1);
    v16bf k01 = load_frag(Kp + (size_t)kb * 64 + 32, 64);
    v16bf k11 = load_frag(Kp + (size_t)(kb + 16) * 64 + 32, 64);
    s0 = WMMA_BF16(qf1, k01, s0);
    s1 = WMMA_BF16(qf1, k11, s1);
    s0 *= 0.125f;
    s1 *= 0.125f;

    // Online softmax per row (row = v + 8*half lives on 16 lanes of a half)
#pragma unroll
    for (int v = 0; v < 8; ++v) {
      float mx = fmaxf(s0[v], s1[v]);
#pragma unroll
      for (int off = 8; off >= 1; off >>= 1)
        mx = fmaxf(mx, __shfl_xor(mx, off, 16));
      float mn = fmaxf(mrow[v], mx);
      float corr = __expf(mrow[v] - mn);
      float p0 = __expf(s0[v] - mn);
      float p1 = __expf(s1[v] - mn);
      float sum = p0 + p1;
#pragma unroll
      for (int off = 8; off >= 1; off >>= 1)
        sum += __shfl_xor(sum, off, 16);
      lrow[v] = lrow[v] * corr + sum;
      mrow[v] = mn;
      o0[v] *= corr; o1[v] *= corr; o2[v] *= corr; o3[v] *= corr;
      // transpose P (D layout -> A layout) through LDS
      int r = v + (half << 3);
      pTile[wv][r][lane & 15] = f2bf(p0);
      pTile[wv][r][16 + (lane & 15)] = f2bf(p1);
    }
    __syncthreads();

    v16bf pf = load_frag(&pTile[wv][0][0], 40);  // 16 q-rows x 32 keys
    // O += P @ V : Vt rows are d-columns -> B fragments load contiguously
    v16bf vf0 = load_frag(Vp + (size_t)0  * 4096 + kb, 4096);
    v16bf vf1 = load_frag(Vp + (size_t)16 * 4096 + kb, 4096);
    v16bf vf2 = load_frag(Vp + (size_t)32 * 4096 + kb, 4096);
    v16bf vf3 = load_frag(Vp + (size_t)48 * 4096 + kb, 4096);
    o0 = WMMA_BF16(pf, vf0, o0);
    o1 = WMMA_BF16(pf, vf1, o1);
    o2 = WMMA_BF16(pf, vf2, o2);
    o3 = WMMA_BF16(pf, vf3, o3);
    __syncthreads();
  }

  const int b = bh >> 3, h = bh & 7;
#pragma unroll
  for (int v = 0; v < 8; ++v) {
    float inv = 1.0f / lrow[v];
    int tok = (b << 12) + q0 + v + (half << 3);
    size_t rowoff = (size_t)tok * 512 + (h << 6) + (lane & 15);
    attnb[rowoff + 0]  = f2bf(o0[v] * inv);
    attnb[rowoff + 16] = f2bf(o1[v] * inv);
    attnb[rowoff + 32] = f2bf(o2[v] * inv);
    attnb[rowoff + 48] = f2bf(o3[v] * inv);
  }
}

// ---------------------------------------------------------------------------
// Output projection: out[8192,512] = attn_bf16 @ Wout + b, fp32 result.
__global__ void __launch_bounds__(128)
gemm_out(const unsigned short* __restrict__ A,   // attn bf16 [8192][512]
         const unsigned short* __restrict__ Bt,  // WoutT bf16 [512][512]
         const float* __restrict__ bias, float* __restrict__ out) {
  const int K = 512;
  const int lane = threadIdx.x & 31;
  const int wv = threadIdx.x >> 5;
  const int m0 = blockIdx.y * 64 + (wv >> 1) * 32;
  const int n0 = blockIdx.x * 64 + (wv & 1) * 32;

  v8f c00 = {}, c01 = {}, c10 = {}, c11 = {};
  for (int kt = 0; kt < K; kt += 32) {
    v16bf a0 = load_frag(A + (size_t)m0 * K + kt, K);
    v16bf a1 = load_frag(A + (size_t)(m0 + 16) * K + kt, K);
    v16bf b0 = load_frag(Bt + (size_t)n0 * K + kt, K);
    v16bf b1 = load_frag(Bt + (size_t)(n0 + 16) * K + kt, K);
    c00 = WMMA_BF16(a0, b0, c00);
    c01 = WMMA_BF16(a0, b1, c01);
    c10 = WMMA_BF16(a1, b0, c10);
    c11 = WMMA_BF16(a1, b1, c11);
  }

  const int half = lane >> 4;
#pragma unroll
  for (int cr = 0; cr < 2; ++cr) {
#pragma unroll
    for (int cc = 0; cc < 2; ++cc) {
      const v8f& c = cr ? (cc ? c11 : c10) : (cc ? c01 : c00);
      int ng = n0 + cc * 16 + (lane & 15);
      float bv = bias[ng];
#pragma unroll
      for (int v = 0; v < 8; ++v) {
        int mg = m0 + cr * 16 + v + (half << 3);
        out[(size_t)mg * 512 + ng] = c[v] + bv;
      }
    }
  }
}

// ---------------------------------------------------------------------------
extern "C" void kernel_launch(void* const* d_in, const int* in_sizes, int n_in,
                              void* d_out, int out_size, void* d_ws, size_t ws_size,
                              hipStream_t stream) {
  (void)in_sizes; (void)n_in; (void)out_size; (void)ws_size;
  const float* x     = (const float*)d_in[0];
  const float* w_qkv = (const float*)d_in[1];
  const float* b_qkv = (const float*)d_in[2];
  const float* w_out = (const float*)d_in[3];
  const float* b_out = (const float*)d_in[4];

  // Workspace layout (bf16 as unsigned short)
  unsigned short* xb    = (unsigned short*)d_ws;   // 8192*512
  unsigned short* wqkvT = xb    + 4194304;         // 1536*512
  unsigned short* woutT = wqkvT + 786432;          // 512*512
  unsigned short* Qb    = woutT + 262144;          // [2,8,4096,64]
  unsigned short* Kb    = Qb    + 4194304;         // [2,8,4096,64]
  unsigned short* Vt    = Kb    + 4194304;         // [2,8,64,4096]
  unsigned short* attnb = Vt    + 4194304;         // [8192,512]

  cvt_bf16<<<4194304 / 256, 256, 0, stream>>>(x, xb, 4194304);
  cvt_bf16_t<<<(786432 + 255) / 256, 256, 0, stream>>>(w_qkv, wqkvT, 512, 1536);
  cvt_bf16_t<<<(262144 + 255) / 256, 256, 0, stream>>>(w_out, woutT, 512, 512);

  gemm_qkv<<<dim3(24, 128), 128, 0, stream>>>(xb, wqkvT, b_qkv, Qb, Kb, Vt);
  flash_attn<<<dim3(64, 16), 128, 0, stream>>>(Qb, Kb, Vt, attnb);
  gemm_out<<<dim3(8, 128), 128, 0, stream>>>(attnb, woutT, b_out, (float*)d_out);
}
